// GRUEncoder_69982197121704
// MI455X (gfx1250) — matrix operand order repaired and checked by reading, not compiled, not run on hardware
//
#include <hip/hip_runtime.h>

#define B_  128
#define S_  1024
#define F_  512
#define H_  128
#define G_  384   // 3*H

typedef __attribute__((ext_vector_type(16))) _Float16 v16h;
typedef __attribute__((ext_vector_type(8)))  _Float16 v8h;
typedef __attribute__((ext_vector_type(8)))  float    v8f;

static __device__ __forceinline__ float sigmoid_f(float x) {
    return 1.0f / (1.0f + __expf(-x));
}
static __device__ __forceinline__ float tanh_f(float x) {
    // stable: x->+inf => 1, x->-inf => -1
    return 2.0f / (1.0f + __expf(-2.0f * x)) - 1.0f;
}

// ---- CDNA5 async copy (ASYNCcnt path, ISA 08_async_tensor §4) ---------------
// LDS address taken from the lds_off VGPR plus the instruction offset; global addr from 64-bit VGPR pair.
static __device__ __forceinline__ void async_load_b128(unsigned lds_off, const void* gptr) {
    asm volatile("global_load_async_to_lds_b128 %0, %1, off"
                 :: "v"(lds_off), "v"(gptr)
                 : "memory");
}
static __device__ __forceinline__ void wait_asynccnt0() {
    asm volatile("s_wait_asynccnt 0" ::: "memory");
}
// Low 32 bits of a flat pointer to __shared__ == wave-relative LDS offset.
static __device__ __forceinline__ unsigned lds_offset_of(const void* p) {
    return (unsigned)(unsigned long long)p;
}

// ---- WMMA fragment loaders (layouts per CDNA5 ISA 7.12.2) -------------------
// A (16x32 f16): lane L holds row M=L%16; halves 0..7 = K kb..kb+7, 8..15 = K kb+16..kb+23
// where kb = 0 (lanes 0-15) or 8 (lanes 16-31). Two contiguous 16B runs.
static __device__ __forceinline__ v16h load_a_f16(const _Float16* base, int ld,
                                                  int row, int k0, int lane) {
    int kb = (lane < 16) ? 0 : 8;
    const _Float16* p = base + (size_t)row * ld + k0 + kb;
    v8h lo = *(const v8h*)p;
    v8h hi = *(const v8h*)(p + 16);
    v16h a;
#pragma unroll
    for (int j = 0; j < 8; ++j) { a[j] = lo[j]; a[8 + j] = hi[j]; }
    return a;
}

// Same A fragment but sourced from fp32 memory with in-flight cvt to f16.
static __device__ __forceinline__ v16h load_a_f32(const float* rowp, int k0, int lane) {
    int kb = (lane < 16) ? 0 : 8;
    const float* p = rowp + k0 + kb;
    float4 f0 = *(const float4*)(p);
    float4 f1 = *(const float4*)(p + 4);
    float4 f2 = *(const float4*)(p + 16);
    float4 f3 = *(const float4*)(p + 20);
    v16h a;
    a[0]=(_Float16)f0.x; a[1]=(_Float16)f0.y; a[2]=(_Float16)f0.z; a[3]=(_Float16)f0.w;
    a[4]=(_Float16)f1.x; a[5]=(_Float16)f1.y; a[6]=(_Float16)f1.z; a[7]=(_Float16)f1.w;
    a[8]=(_Float16)f2.x; a[9]=(_Float16)f2.y; a[10]=(_Float16)f2.z; a[11]=(_Float16)f2.w;
    a[12]=(_Float16)f3.x; a[13]=(_Float16)f3.y; a[14]=(_Float16)f3.z; a[15]=(_Float16)f3.w;
    return a;
}

// B (32x16 f16): lane holds column ncol; halves 0..7 = K kb..kb+7, 8..15 = K kb+16..kb+23
// (kb = 0 lanes 0-15, 8 lanes 16-31). Weights stored [N,K] row-major, so
// a lane's fragment is two contiguous 16-byte loads.
static __device__ __forceinline__ v16h load_b_f16(const _Float16* w, int ld,
                                                  int ncol, int k0, int lane) {
    int kb = (lane < 16) ? 0 : 8;
    const _Float16* p = w + (size_t)ncol * ld + k0 + kb;
    v8h lo = *(const v8h*)p;
    v8h hi = *(const v8h*)(p + 16);
    v16h b;
#pragma unroll
    for (int j = 0; j < 8; ++j) { b[j] = lo[j]; b[8 + j] = hi[j]; }
    return b;
}

// ---- Stage 1: valid lengths per batch row ----------------------------------
// One wave per (b,t) row; wave-reduce the 512-float feature sum.
__global__ void k_lengths(const float* __restrict__ x, int* __restrict__ lengths) {
    int wave = threadIdx.x >> 5;
    int lane = threadIdx.x & 31;
    int row  = blockIdx.x * 8 + wave;                 // row = b*S + t
    const float* p = x + (size_t)row * F_ + lane * 16;
    float s = 0.f;
#pragma unroll
    for (int j = 0; j < 4; ++j) {
        float4 v = *(const float4*)(p + j * 4);
        s += v.x + v.y + v.z + v.w;
    }
#pragma unroll
    for (int off = 16; off > 0; off >>= 1) s += __shfl_xor(s, off, 32);
    if (lane == 0 && s != 0.f) atomicAdd(&lengths[row / S_], 1);
}

// ---- Stage 2: stable descending argsort via rank counting (B=128) ----------
__global__ void k_order(const int* __restrict__ lengths, int* __restrict__ order,
                        int* __restrict__ slen) {
    int b  = threadIdx.x;
    int lb = lengths[b];
    int rank = 0;
    for (int j = 0; j < B_; ++j) {
        int lj = lengths[j];
        rank += (lj > lb) || (lj == lb && j < b);
    }
    order[rank] = b;
    slen[rank]  = lb;
}

// ---- Stage 3: fp32 -> f16 weight conversion --------------------------------
__global__ void k_cvt_f16(const float* __restrict__ src, _Float16* __restrict__ dst, int n) {
    int i = blockIdx.x * blockDim.x + threadIdx.x;
    if (i < n) dst[i] = (_Float16)src[i];
}

// ---- Stage 4: projection GEMM + batch reorder + transpose ------------------
// proj[t, i, h] = f16( x[order[i], t, :] . Wp[h, :] + bp[h] )
// grid (S/16, B), block 256 = 8 waves; wave w owns N-tile w (16 of 128 h).
__global__ void k_proj(const float* __restrict__ x, const _Float16* __restrict__ wp16,
                       const float* __restrict__ bp, const int* __restrict__ order,
                       _Float16* __restrict__ proj) {
    int lane = threadIdx.x & 31;
    int wv   = threadIdx.x >> 5;
    int t0   = blockIdx.x * 16;
    int bi   = blockIdx.y;                    // sorted position
    int bsrc = order[bi];
    int m    = lane & 15;
    int n_g  = wv * 16 + m;
    const float* arow = x + ((size_t)bsrc * S_ + t0 + m) * F_;
    v8f acc = {};
#pragma unroll 4
    for (int k0 = 0; k0 < F_; k0 += 32) {
        v16h a = load_a_f32(arow, k0, lane);
        v16h b = load_b_f16(wp16, F_, n_g, k0, lane);
        acc = __builtin_amdgcn_wmma_f32_16x16x32_f16(false, a, false, b,
                                                     (short)0, acc, false, false);
    }
    float bias = bp[n_g];
    int mo = (lane < 16) ? 0 : 8;
#pragma unroll
    for (int r = 0; r < 8; ++r) {
        int t = t0 + r + mo;
        proj[((size_t)t * B_ + bi) * H_ + n_g] = (_Float16)(acc[r] + bias);
    }
}

// ---- Stage 5: input-gate GEMM  xg = inp @ Wih^T + bih ----------------------
// inp: [S*B, 128] f16 row-major; wih: [384,128] f16; xg: [S*B, 384] f16.
// grid (S*B/16, 3), block 256; wave handles N-tile blockIdx.y*8 + wave.
__global__ void k_gates(const _Float16* __restrict__ inp, const _Float16* __restrict__ wih,
                        const float* __restrict__ bih, _Float16* __restrict__ xg) {
    int lane = threadIdx.x & 31;
    int wv   = threadIdx.x >> 5;
    int row0 = blockIdx.x * 16;
    int nt   = blockIdx.y * 8 + wv;           // 0..23
    int m    = lane & 15;
    int n_g  = nt * 16 + m;
    v8f acc = {};
#pragma unroll
    for (int k0 = 0; k0 < H_; k0 += 32) {
        v16h a = load_a_f16(inp, H_, row0 + m, k0, lane);
        v16h b = load_b_f16(wih, H_, n_g, k0, lane);
        acc = __builtin_amdgcn_wmma_f32_16x16x32_f16(false, a, false, b,
                                                     (short)0, acc, false, false);
    }
    float bias = bih[n_g];
    int mo = (lane < 16) ? 0 : 8;
#pragma unroll
    for (int r = 0; r < 8; ++r)
        xg[(size_t)(row0 + r + mo) * G_ + n_g] = (_Float16)(acc[r] + bias);
}

// ---- Stage 6: recurrent layer ----------------------------------------------
// 8 workgroups, each owns 16 batch rows (GRU recurrence is independent per
// batch row -> no cross-WGP sync). h lives in LDS (16x128 f16). Wave w owns
// hidden units [16w,16w+16): gate tiles w (r), w+8 (z), w+16 (n), so the
// nonlinear gate combine is register-local. Whh B-fragments (12 x v16h) are
// preloaded into VGPRs and reused for all 1024 steps. The xg gate tile for
// step t+1 (one contiguous 12KB span: rows t*B+i0..+15) is DMA'd into a
// double-buffered LDS stage with GLOBAL_LOAD_ASYNC_TO_LDS_B128, overlapping
// the whole step-t compute; each wave waits its own ASYNCcnt just before the
// end-of-step barrier.
template <bool LAST>
__global__ void k_rec(const _Float16* __restrict__ xg, const _Float16* __restrict__ whh,
                      const float* __restrict__ bhh, const int* __restrict__ slen,
                      _Float16* __restrict__ hseq, float* __restrict__ out) {
    __shared__ _Float16 hbuf[16 * H_];
    __shared__ _Float16 xgbuf[2][16 * G_];            // 2 x 12KB double buffer
    int lane = threadIdx.x & 31;
    int wv   = threadIdx.x >> 5;              // 0..7
    int i0   = blockIdx.x * 16;               // batch-row block (sorted order)
    int m    = lane & 15;
    int mo   = (lane < 16) ? 0 : 8;
    int hu   = wv * 16 + m;                   // hidden unit (C-layout column)
    int toff = threadIdx.x * 16;              // this thread's byte slice of a tile

    // prologue: kick off DMA of tile 0 while we set up
    {
        const char* g0 = (const char*)(xg + ((size_t)0 * B_ + i0) * G_);
        unsigned l0 = lds_offset_of(&xgbuf[0][0]) + toff;
#pragma unroll
        for (int c = 0; c < 3; ++c)
            async_load_b128(l0 + c * 4096, g0 + toff + c * 4096);
    }

    for (int idx = threadIdx.x; idx < 16 * H_; idx += blockDim.x)
        hbuf[idx] = (_Float16)0.f;

    v16h br[4], bz[4], bn[4];
#pragma unroll
    for (int c = 0; c < 4; ++c) {
        br[c] = load_b_f16(whh, H_, hu,          c * 32, lane);
        bz[c] = load_b_f16(whh, H_, H_ + hu,     c * 32, lane);
        bn[c] = load_b_f16(whh, H_, 2 * H_ + hu, c * 32, lane);
    }
    float bhr = bhh[hu], bhz = bhh[H_ + hu], bhn = bhh[2 * H_ + hu];

    int   lenr[8];
    float accum[8];
#pragma unroll
    for (int r = 0; r < 8; ++r) { lenr[r] = slen[i0 + r + mo]; accum[r] = 0.f; }

    wait_asynccnt0();
    __syncthreads();                                   // tile 0 resident, h zeroed

    for (int t = 0; t < S_; ++t) {
        int cur = t & 1, nxt = cur ^ 1;

        // kick off DMA of tile t+1 into the other buffer (its last readers
        // finished before the previous step's mid barrier).
        if (t + 1 < S_) {
            const char* gn = (const char*)(xg + ((size_t)(t + 1) * B_ + i0) * G_);
            unsigned ln = lds_offset_of(&xgbuf[nxt][0]) + toff;
#pragma unroll
            for (int c = 0; c < 3; ++c)
                async_load_b128(ln + c * 4096, gn + toff + c * 4096);
        }

        // Init accumulators from the staged xg tile: r,z get xg+bhh folded in;
        // n keeps hg separate because n = tanh(xn + r*(h@Whh_n^T + bhh_n)).
        v8f ar, az, an;
        float xn[8];
#pragma unroll
        for (int r = 0; r < 8; ++r) {
            const _Float16* p = &xgbuf[cur][(r + mo) * G_];
            ar[r] = (float)p[hu] + bhr;
            az[r] = (float)p[H_ + hu] + bhz;
            xn[r] = (float)p[2 * H_ + hu];
            an[r] = bhn;
        }
        // hg += h @ Whh^T : A fragments from LDS h, B fragments resident in VGPRs
#pragma unroll
        for (int c = 0; c < 4; ++c) {
            v16h a = load_a_f16(hbuf, H_, m, c * 32, lane);
            ar = __builtin_amdgcn_wmma_f32_16x16x32_f16(false, a, false, br[c],
                                                        (short)0, ar, false, false);
            az = __builtin_amdgcn_wmma_f32_16x16x32_f16(false, a, false, bz[c],
                                                        (short)0, az, false, false);
            an = __builtin_amdgcn_wmma_f32_16x16x32_f16(false, a, false, bn[c],
                                                        (short)0, an, false, false);
        }
        float hold[8];
#pragma unroll
        for (int r = 0; r < 8; ++r) hold[r] = (float)hbuf[(r + mo) * H_ + hu];
        __syncthreads();                       // all h / xg[cur] reads complete
#pragma unroll
        for (int r = 0; r < 8; ++r) {
            float rg = sigmoid_f(ar[r]);
            float zg = sigmoid_f(az[r]);
            float ng = tanh_f(xn[r] + rg * an[r]);
            float hn = (1.f - zg) * ng + zg * hold[r];
            hbuf[(r + mo) * H_ + hu] = (_Float16)hn;
            if (!LAST) {
                hseq[((size_t)t * B_ + i0 + r + mo) * H_ + hu] = (_Float16)hn;
            } else {
                accum[r] += (t < lenr[r]) ? hn : 0.f;
            }
        }
        wait_asynccnt0();                      // my slice of tile t+1 landed
        __syncthreads();                       // all slices landed, h writes visible
    }
    if (LAST) {
#pragma unroll
        for (int r = 0; r < 8; ++r)
            out[(size_t)(i0 + r + mo) * H_ + hu] = accum[r] * (1.0f / (float)S_);
    }
}

extern "C" void kernel_launch(void* const* d_in, const int* in_sizes, int n_in,
                              void* d_out, int out_size, void* d_ws, size_t ws_size,
                              hipStream_t stream) {
    (void)in_sizes; (void)n_in; (void)out_size; (void)ws_size;
    const float* x    = (const float*)d_in[0];
    const float* Wp   = (const float*)d_in[1];
    const float* bp   = (const float*)d_in[2];
    const float* Wih0 = (const float*)d_in[3];
    const float* Whh0 = (const float*)d_in[4];
    const float* bih0 = (const float*)d_in[5];
    const float* bhh0 = (const float*)d_in[6];
    const float* Wih1 = (const float*)d_in[7];
    const float* Whh1 = (const float*)d_in[8];
    const float* bih1 = (const float*)d_in[9];
    const float* bhh1 = (const float*)d_in[10];
    float* out = (float*)d_out;

    char* ws = (char*)d_ws;
    size_t o = 0;
    int* lengths = (int*)(ws + o); o += 128 * sizeof(int);
    int* order   = (int*)(ws + o); o += 128 * sizeof(int);
    int* slen    = (int*)(ws + o); o += 128 * sizeof(int);
    o = (o + 255) & ~(size_t)255;
    _Float16* wp16   = (_Float16*)(ws + o); o += (size_t)H_ * F_ * 2;
    _Float16* wih0_h = (_Float16*)(ws + o); o += (size_t)G_ * H_ * 2;
    _Float16* whh0_h = (_Float16*)(ws + o); o += (size_t)G_ * H_ * 2;
    _Float16* wih1_h = (_Float16*)(ws + o); o += (size_t)G_ * H_ * 2;
    _Float16* whh1_h = (_Float16*)(ws + o); o += (size_t)G_ * H_ * 2;
    o = (o + 255) & ~(size_t)255;
    _Float16* proj16 = (_Float16*)(ws + o); o += (size_t)S_ * B_ * H_ * 2;  // reused as h1
    _Float16* xg16   = (_Float16*)(ws + o); o += (size_t)S_ * B_ * G_ * 2;  // reused by layer1

    hipMemsetAsync(lengths, 0, 128 * sizeof(int), stream);
    k_lengths<<<B_ * S_ / 8, 256, 0, stream>>>(x, lengths);
    k_order<<<1, 128, 0, stream>>>(lengths, order, slen);

    int nW = H_ * F_, nG = G_ * H_;
    k_cvt_f16<<<(nW + 255) / 256, 256, 0, stream>>>(Wp,   wp16,   nW);
    k_cvt_f16<<<(nG + 255) / 256, 256, 0, stream>>>(Wih0, wih0_h, nG);
    k_cvt_f16<<<(nG + 255) / 256, 256, 0, stream>>>(Whh0, whh0_h, nG);
    k_cvt_f16<<<(nG + 255) / 256, 256, 0, stream>>>(Wih1, wih1_h, nG);
    k_cvt_f16<<<(nG + 255) / 256, 256, 0, stream>>>(Whh1, whh1_h, nG);

    k_proj<<<dim3(S_ / 16, B_), 256, 0, stream>>>(x, wp16, bp, order, proj16);

    k_gates<<<dim3(S_ * B_ / 16, 3), 256, 0, stream>>>(proj16, wih0_h, bih0, xg16);
    k_rec<false><<<B_ / 16, 256, 0, stream>>>(xg16, whh0_h, bhh0, slen, proj16, out);

    k_gates<<<dim3(S_ * B_ / 16, 3), 256, 0, stream>>>(proj16, wih1_h, bih1, xg16);
    k_rec<true><<<B_ / 16, 256, 0, stream>>>(xg16, whh1_h, bhh1, slen, proj16, out);
}
